// WindowAttention1D_13786845020191
// MI455X (gfx1250) — compile-verified, matrix operand order checked by reading, not run on hardware
//
#include <hip/hip_runtime.h>

// ---------------------------------------------------------------------------
// WindowAttention1D for MI455X (gfx1250, wave32, WMMA + TDM)
// B=32, S=4096, D=512, H=8, hd=64, window=128, shift=64
// Pipeline:
//   [roll+cvt x->bf16] [cvt weights->bf16]       (memory-bound pre-passes)
//   [QKV GEMM]  -> [per-(win,head) attention] -> [out GEMM + unroll]
// Matrix math: v_wmma_f32_16x16x32_bf16, f32 accumulate.
// B operands staged in LDS via TENSOR_LOAD_TO_LDS (one DMA per block,
// hardware LDS padding for conflict-free ds_load_b128), A streams from HBM.
// ---------------------------------------------------------------------------

typedef __attribute__((ext_vector_type(16))) __bf16 v16bf;
typedef __attribute__((ext_vector_type(8)))  float  v8f;
typedef __attribute__((ext_vector_type(4)))  unsigned int v4u;
typedef __attribute__((ext_vector_type(8)))  int v8i;
typedef __attribute__((ext_vector_type(4)))  int v4i;

union BF16x16 {
  v16bf v;
  unsigned short u[16];
  uint4 q4[2];
};

#define S_LEN  4096
#define D_DIM  512
#define HEADS  8
#define HD     64
#define WSZ    128
#define NWIN   1024          // B * (S / WSZ)
#define TOKS   (32 * 4096)   // 131072 token rows

// padded LDS row strides (bf16 elems): +8 elems (4 dwords) per row
#define BSTR   520           // 512-elem rows  (1040B = 65*16B, 4-bank rotate)
#define KSTR   72            // 64-elem rows
#define VSTR   136           // 128-elem rows

// fp32 -> bf16 RNE (scalar)
__device__ __forceinline__ unsigned short f2bf(float f) {
  unsigned int u = __float_as_uint(f);
  u += 0x7FFFu + ((u >> 16) & 1u);
  return (unsigned short)(u >> 16);
}

// pack two fp32 -> one dword of two bf16 (RNE)
__device__ __forceinline__ unsigned int pk2(float lo, float hi) {
  unsigned int a = __float_as_uint(lo);
  unsigned int b = __float_as_uint(hi);
  a += 0x7FFFu + ((a >> 16) & 1u);
  b += 0x7FFFu + ((b >> 16) & 1u);
  return (a >> 16) | (b & 0xFFFF0000u);
}

__device__ __forceinline__ v8f wmma_bf16(const BF16x16& A, const BF16x16& B, v8f C) {
  return __builtin_amdgcn_wmma_f32_16x16x32_bf16(false, A.v, false, B.v,
                                                 (short)0, C, false, false);
}

// load one 16-lane-striped bf16 operand tile (two b128 loads, global or LDS)
__device__ __forceinline__ void ld_tile(BF16x16& T, const unsigned short* p) {
  T.q4[0] = *(const uint4*)(p);
  T.q4[1] = *(const uint4*)(p + 16);
}

// ---------------------------------------------------------------------------
// TDM: 1-D contiguous bf16 copy global->LDS with hardware LDS row padding.
//   nelem  : number of bf16 elements (<= 65535, tile_dim0 is 16-bit)
//   interval: pad_interval code (4: every 32 dw / 64 elems, 5: every 64 dw,
//             7: every 256 dw / 512 elems) ; pad_amount = 4 dwords (16B)
// Descriptor per CDNA5 ISA 8.3/8.4: count=1, type=2("image"), data_size=2B.
// ---------------------------------------------------------------------------
__device__ __forceinline__ void tdm_load_1d(unsigned lds_addr,
                                            const unsigned short* gptr,
                                            unsigned nelem, unsigned interval) {
  const unsigned long long ga = (unsigned long long)(const void*)gptr;
  v4u g0;
  g0.x = 1u;                                   // count=1 (valid user D#)
  g0.y = lds_addr;                             // LDS byte address
  g0.z = (unsigned)ga;                         // global_addr[31:0]
  g0.w = (unsigned)(ga >> 32) | (2u << 30);    // global_addr[56:32] | type=2
  v8i g1;
  g1[0] = (int)((1u << 16) |                   // data_size = 2 bytes
                (1u << 20) |                   // pad_enable (LDS dest)
                (interval << 22) |             // pad_interval
                (3u << 25));                   // pad_amount = 4 dwords (16B)
  g1[1] = (int)((nelem & 0xFFFFu) << 16);      // tensor_dim0[15:0]
  g1[2] = (int)((nelem >> 16) & 0xFFFFu) | (1 << 16); // dim0[31:16] | tensor_dim1=1
  g1[3] = (int)((nelem & 0xFFFFu) << 16);      // tile_dim0
  g1[4] = 1;                                   // tile_dim1 = 1
  g1[5] = (int)nelem;                          // tensor_dim0_stride[31:0]
  g1[6] = 0;
  g1[7] = 0;
  v4i z4 = {0, 0, 0, 0};
#if __clang_major__ >= 23
  v8i z8 = {0, 0, 0, 0, 0, 0, 0, 0};
  __builtin_amdgcn_tensor_load_to_lds(g0, g1, z4, z4, z8, 0);
#else
  __builtin_amdgcn_tensor_load_to_lds(g0, g1, z4, z4, 0);
#endif
}

__device__ __forceinline__ unsigned lds_off(const void* p) {
  return (unsigned)(unsigned long long)p;   // flat LDS addr: low 32b = offset
}

// ---------------------------------------------------------------------------
// Pre-pass: x (fp32) -> xb (bf16) with roll(-shift) folded into source row.
// ---------------------------------------------------------------------------
__global__ __launch_bounds__(256)
void roll_cvt_kernel(const float* __restrict__ x, const int* __restrict__ shift_p,
                     unsigned short* __restrict__ xb) {
  const int tid = blockIdx.x * 256 + threadIdx.x;   // TOKS * 64 threads
  const int row = tid >> 6;
  const int c   = (tid & 63) * 8;
  const int b   = row >> 12;
  const int s_r = row & (S_LEN - 1);
  const int s_in = (s_r + *shift_p) & (S_LEN - 1);
  const float4* p = (const float4*)(x + (size_t)(b * S_LEN + s_in) * D_DIM + c);
  float4 a0 = p[0], a1 = p[1];
  uint4 o;
  o.x = pk2(a0.x, a0.y); o.y = pk2(a0.z, a0.w);
  o.z = pk2(a1.x, a1.y); o.w = pk2(a1.z, a1.w);
  *(uint4*)(xb + (size_t)row * D_DIM + c) = o;
}

// ---------------------------------------------------------------------------
// Pre-pass: dense fp32 -> bf16 (weights).
// ---------------------------------------------------------------------------
__global__ __launch_bounds__(256)
void cvt_w_kernel(const float* __restrict__ src, unsigned short* __restrict__ dst,
                  int n8) {
  const int i = blockIdx.x * 256 + threadIdx.x;
  if (i >= n8) return;
  const float4* p = (const float4*)src + (size_t)2 * i;
  float4 a0 = p[0], a1 = p[1];
  uint4 o;
  o.x = pk2(a0.x, a0.y); o.y = pk2(a0.z, a0.w);
  o.z = pk2(a1.x, a1.y); o.w = pk2(a1.z, a1.w);
  ((uint4*)dst)[i] = o;
}

// ---------------------------------------------------------------------------
// Kernel 1: qkv = xb @ wb_in^T + b_in.  B panel (64 rows x 512 K) TDM-staged
// in LDS once per block; A streams from global. wave = 16x64 (4 accs).
// ---------------------------------------------------------------------------
__global__ __launch_bounds__(256)
void qkv_kernel(const unsigned short* __restrict__ xb,
                const unsigned short* __restrict__ wb,
                const float* __restrict__ b_in,
                unsigned short* __restrict__ qb, unsigned short* __restrict__ kb,
                unsigned short* __restrict__ vb) {
  __shared__ unsigned short Bs[64 * BSTR];   // 66,560 B

  const int lane = threadIdx.x & 31;
  const int wv   = threadIdx.x >> 5;
  const int l15  = lane & 15;
  const int hihalf = (lane >> 4) & 1;
  const int koff = hihalf * 8;

  const int colBase = blockIdx.x * 64;            // 0..1535 (q|k|v packed)
  const int rowBase = blockIdx.y * 128 + wv * 16; // token rows

  if (wv == 0) {                                  // one DMA per block
    tdm_load_1d(lds_off(&Bs[0]), wb + (size_t)colBase * D_DIM,
                64 * D_DIM, /*pad every 512 elems*/ 7u);
    __builtin_amdgcn_s_wait_tensorcnt(0);
  }
  __syncthreads();

  const unsigned short* arow = xb + (size_t)(rowBase + l15) * D_DIM + koff;
  __builtin_prefetch(arow, 0, 3);                 // global_prefetch_b8
  __builtin_prefetch(arow + 512, 0, 3);

  const unsigned short* brow[4];
#pragma unroll
  for (int j = 0; j < 4; ++j)
    brow[j] = &Bs[(16 * j + l15) * BSTR + koff];

  v8f acc[4] = {};
#pragma unroll
  for (int k0 = 0; k0 < D_DIM; k0 += 32) {
    BF16x16 A;
    ld_tile(A, arow + k0);                        // global b128 x2
#pragma unroll
    for (int j = 0; j < 4; ++j) {
      BF16x16 Bm;
      ld_tile(Bm, brow[j] + k0);                  // ds_load_b128 x2
      acc[j] = wmma_bf16(A, Bm, acc[j]);
    }
  }

  // Epilogue: +bias, scatter q/k head-major, v transposed.
  const int which = colBase >> 9;        // 0=q 1=k 2=v
  const int hd_h  = (colBase & 511) >> 6;

#pragma unroll
  for (int j = 0; j < 4; ++j) {
    const int c    = colBase + 16 * j + l15;
    const float bias = b_in[c];
    const int d    = 16 * j + l15;
#pragma unroll
    for (int r = 0; r < 8; ++r) {
      const int m   = rowBase + r + hihalf * 8;
      const int win = m >> 7;
      const int t   = m & 127;
      const unsigned short bv = f2bf(acc[j][r] + bias);
      const size_t wh = (size_t)(win * HEADS + hd_h);
      if (which == 0)      qb[(wh * WSZ + t) * HD + d] = bv;
      else if (which == 1) kb[(wh * WSZ + t) * HD + d] = bv;
      else                 vb[(wh * HD + d) * WSZ + t] = bv;
    }
  }
}

// ---------------------------------------------------------------------------
// Kernel 2: per-(window, head) attention. K and V^T tiles TDM-staged in LDS.
// ---------------------------------------------------------------------------
__global__ __launch_bounds__(256)
void attn_kernel(const unsigned short* __restrict__ qb,
                 const unsigned short* __restrict__ kb,
                 const unsigned short* __restrict__ vb,
                 unsigned short* __restrict__ ob) {
  __shared__ unsigned short Ks[WSZ * KSTR];   // 18,432 B (128 rows x 64 +pad)
  __shared__ unsigned short Vs[HD * VSTR];    // 17,408 B (64 rows x 128 +pad)
  __shared__ unsigned short Ps[WSZ * VSTR];   // 34,816 B (128 rows x 128 +pad)

  const int lane = threadIdx.x & 31;
  const int wv   = threadIdx.x >> 5;
  const int l15  = lane & 15;
  const int hihalf = (lane >> 4) & 1;
  const int koff = hihalf * 8;

  const int win = blockIdx.x >> 3;
  const int h   = blockIdx.x & 7;
  const unsigned short* qh = qb + (size_t)(win * HEADS + h) * WSZ * HD;
  const unsigned short* kh = kb + (size_t)(win * HEADS + h) * WSZ * HD;
  const unsigned short* vh = vb + (size_t)(win * HEADS + h) * HD * WSZ;

  if (wv == 0) {
    tdm_load_1d(lds_off(&Ks[0]), kh, WSZ * HD, /*pad every 64 elems*/ 4u);
    tdm_load_1d(lds_off(&Vs[0]), vh, HD * WSZ, /*pad every 128 elems*/ 5u);
    __builtin_amdgcn_s_wait_tensorcnt(0);
  }
  __syncthreads();

  // ---- scores = q @ k^T (A: global q rows; B: LDS K rows) ---------------
  v8f sacc[8] = {};
  const unsigned short* qrow = qh + (size_t)(16 * wv + l15) * HD + koff;
#pragma unroll
  for (int k0 = 0; k0 < HD; k0 += 32) {
    BF16x16 A;
    ld_tile(A, qrow + k0);
#pragma unroll
    for (int j = 0; j < 8; ++j) {
      BF16x16 Bm;
      ld_tile(Bm, &Ks[(16 * j + l15) * KSTR + koff + k0]);
      sacc[j] = wmma_bf16(A, Bm, sacc[j]);
    }
  }

  // ---- softmax over 128 keys per query row ------------------------------
  const float scale = 0.125f;   // 1/sqrt(64)
#pragma unroll
  for (int r = 0; r < 8; ++r) {
    float mx = sacc[0][r];
#pragma unroll
    for (int j = 1; j < 8; ++j) mx = fmaxf(mx, sacc[j][r]);
    mx = fmaxf(mx, __shfl_xor(mx, 1, 32));
    mx = fmaxf(mx, __shfl_xor(mx, 2, 32));
    mx = fmaxf(mx, __shfl_xor(mx, 4, 32));
    mx = fmaxf(mx, __shfl_xor(mx, 8, 32));
    float e[8];
    float sum = 0.f;
#pragma unroll
    for (int j = 0; j < 8; ++j) {
      e[j] = __expf((sacc[j][r] - mx) * scale);
      sum += e[j];
    }
    sum += __shfl_xor(sum, 1, 32);
    sum += __shfl_xor(sum, 2, 32);
    sum += __shfl_xor(sum, 4, 32);
    sum += __shfl_xor(sum, 8, 32);
    const float inv = 1.0f / sum;
    const int row = 16 * wv + r + hihalf * 8;
#pragma unroll
    for (int j = 0; j < 8; ++j)
      Ps[row * VSTR + 16 * j + l15] = f2bf(e[j] * inv);
  }
  __syncthreads();

  // ---- o = P @ V (both operands from LDS) -------------------------------
  v8f oacc[4] = {};
  const unsigned short* prow = &Ps[(16 * wv + l15) * VSTR + koff];
#pragma unroll
  for (int k0 = 0; k0 < WSZ; k0 += 32) {
    BF16x16 A;
    ld_tile(A, prow + k0);
#pragma unroll
    for (int j = 0; j < 4; ++j) {
      BF16x16 Bm;
      ld_tile(Bm, &Vs[(16 * j + l15) * VSTR + koff + k0]);
      oacc[j] = wmma_bf16(A, Bm, oacc[j]);
    }
  }

  // ---- store o (bf16, [token][512], heads concatenated) -----------------
#pragma unroll
  for (int j = 0; j < 4; ++j) {
#pragma unroll
    for (int r = 0; r < 8; ++r) {
      const int t = 16 * wv + r + hihalf * 8;
      const size_t gt = (size_t)win * WSZ + t;
      ob[gt * D_DIM + h * HD + 16 * j + l15] = f2bf(oacc[j][r]);
    }
  }
}

// ---------------------------------------------------------------------------
// Kernel 3: out = o @ wb_out^T + b_out, inverse roll folded into fp32 store.
// B panel TDM-staged in LDS.
// ---------------------------------------------------------------------------
__global__ __launch_bounds__(256)
void proj_kernel(const unsigned short* __restrict__ ob,
                 const unsigned short* __restrict__ wb,
                 const float* __restrict__ b_out,
                 const int* __restrict__ shift_p, float* __restrict__ out) {
  __shared__ unsigned short Bs[64 * BSTR];

  const int lane = threadIdx.x & 31;
  const int wv   = threadIdx.x >> 5;
  const int l15  = lane & 15;
  const int hihalf = (lane >> 4) & 1;
  const int koff = hihalf * 8;
  const int shift = *shift_p;

  const int colBase = blockIdx.x * 64;
  const int rowBase = blockIdx.y * 128 + wv * 16;

  if (wv == 0) {
    tdm_load_1d(lds_off(&Bs[0]), wb + (size_t)colBase * D_DIM,
                64 * D_DIM, 7u);
    __builtin_amdgcn_s_wait_tensorcnt(0);
  }
  __syncthreads();

  const unsigned short* arow = ob + (size_t)(rowBase + l15) * D_DIM + koff;
  __builtin_prefetch(arow, 0, 3);
  __builtin_prefetch(arow + 512, 0, 3);

  const unsigned short* brow[4];
#pragma unroll
  for (int j = 0; j < 4; ++j)
    brow[j] = &Bs[(16 * j + l15) * BSTR + koff];

  v8f acc[4] = {};
#pragma unroll
  for (int k0 = 0; k0 < D_DIM; k0 += 32) {
    BF16x16 A;
    ld_tile(A, arow + k0);
#pragma unroll
    for (int j = 0; j < 4; ++j) {
      BF16x16 Bm;
      ld_tile(Bm, brow[j] + k0);
      acc[j] = wmma_bf16(A, Bm, acc[j]);
    }
  }

#pragma unroll
  for (int j = 0; j < 4; ++j) {
    const int c = colBase + 16 * j + l15;
    const float bias = b_out[c];
#pragma unroll
    for (int r = 0; r < 8; ++r) {
      const int m    = rowBase + r + hihalf * 8;
      const int bidx = m >> 12;
      const int s_r  = m & (S_LEN - 1);
      const int s_o  = (s_r + shift) & (S_LEN - 1);
      out[(size_t)(bidx * S_LEN + s_o) * D_DIM + c] = acc[j][r] + bias;
    }
  }
}

// ---------------------------------------------------------------------------
extern "C" void kernel_launch(void* const* d_in, const int* in_sizes, int n_in,
                              void* d_out, int out_size, void* d_ws, size_t ws_size,
                              hipStream_t stream) {
  (void)in_sizes; (void)n_in; (void)out_size; (void)ws_size;
  const float* x     = (const float*)d_in[0];
  const float* w_in  = (const float*)d_in[1];
  const float* b_in  = (const float*)d_in[2];
  const float* w_out = (const float*)d_in[3];
  const float* b_out = (const float*)d_in[4];
  const int*   shift = (const int*)d_in[6];
  float* out = (float*)d_out;

  const size_t XB_ELEMS  = (size_t)TOKS * D_DIM;            // 67,108,864
  const size_t QKV_ELEMS = (size_t)NWIN * HEADS * WSZ * HD; // 67,108,864
  unsigned short* xb    = (unsigned short*)d_ws;
  unsigned short* qb    = xb + XB_ELEMS;
  unsigned short* kb    = qb + QKV_ELEMS;
  unsigned short* vb    = kb + QKV_ELEMS;
  unsigned short* wbin  = vb + QKV_ELEMS;
  unsigned short* wbout = wbin + (size_t)3 * D_DIM * D_DIM;
  unsigned short* obuf  = xb;   // reuse: xb is dead after qkv_kernel

  dim3 blk(256);
  roll_cvt_kernel<<<dim3(TOKS * 64 / 256), blk, 0, stream>>>(x, shift, xb);
  cvt_w_kernel<<<dim3((3 * D_DIM * D_DIM / 8 + 255) / 256), blk, 0, stream>>>(
      w_in, wbin, 3 * D_DIM * D_DIM / 8);
  cvt_w_kernel<<<dim3((D_DIM * D_DIM / 8 + 255) / 256), blk, 0, stream>>>(
      w_out, wbout, D_DIM * D_DIM / 8);
  qkv_kernel<<<dim3(3 * D_DIM / 64, TOKS / 128), blk, 0, stream>>>(
      xb, wbin, b_in, qb, kb, vb);
  attn_kernel<<<dim3(NWIN * HEADS), blk, 0, stream>>>(qb, kb, vb, obuf);
  proj_kernel<<<dim3(D_DIM / 64, TOKS / 128), blk, 0, stream>>>(
      obuf, wbout, b_out, shift, out);
}
